// MatrixStateElmanCell_70214125355419
// MI455X (gfx1250) — compile-verified, matrix operand order checked by reading, not run on hardware
//
#include <hip/hip_runtime.h>
#include <hip/hip_bf16.h>

// ---------------------------------------------------------------------------
// MatrixStateElmanCell for MI455X (gfx1250, wave32).
//   Stage 1: convert x -> bf16, transpose-convert weights -> bf16 [N,K] layout
//   Stage 2: WMMA bf16 GEMMs (v_wmma_f32_16x16x32_bf16), 2x2 register-blocked
//            tiles per wave (pinned accumulators, no hazard nops), fused
//            bias + fast activation (v_exp_f32 + v_rcp_f32, no IEEE div)
//   Stage 3: register-resident scan over T; H_all (1.08 GB, write-once)
//            streamed with non-temporal stores to keep L2 for val/qry/key/dec
// ---------------------------------------------------------------------------

#define T_STEPS 512
#define BATCH   8
#define D_MODEL 1024
#define D_STATE 64
#define M_ROWS  (T_STEPS * BATCH)   // 4096

typedef __attribute__((ext_vector_type(16))) __bf16 v16bf;
typedef __attribute__((ext_vector_type(8)))  __bf16 v8bf;
typedef __attribute__((ext_vector_type(8)))  float  v8f;
typedef __attribute__((ext_vector_type(2)))  float  v2f;

// Fast transcendentals: single v_exp_f32 / v_rcp_f32, no Newton refinement.
__device__ __forceinline__ float fast_sigmoid(float x) {
    return __builtin_amdgcn_rcpf(1.0f + __expf(-x));
}
__device__ __forceinline__ float fast_tanh(float x) {
    // tanh(x) = 1 - 2/(exp(2x)+1)
    return 1.0f - 2.0f * __builtin_amdgcn_rcpf(1.0f + __expf(2.0f * x));
}

// -------------------------------- Stage 1 ----------------------------------

__global__ void cvt_f32_to_bf16(const float* __restrict__ in,
                                __bf16* __restrict__ out, int n) {
    int i = blockIdx.x * blockDim.x + threadIdx.x;
    if (i < n) out[i] = (__bf16)in[i];
}

// W: [Kd, N] row-major (reference stores [in, out]) -> Wt: [N, Kd] bf16.
__global__ void transpose_cvt_bf16(const float* __restrict__ W,
                                   __bf16* __restrict__ Wt, int Kd, int N) {
    int i = blockIdx.x * blockDim.x + threadIdx.x;
    if (i >= Kd * N) return;
    int n = i / Kd;
    int k = i - n * Kd;
    Wt[i] = (__bf16)W[k * N + n];
}

// -------------------------------- Stage 2 ----------------------------------
// Register-blocked WMMA GEMM: each wave computes an (MT*16) x (NT*16) block.
// A: [M, Kd] bf16 row-major. Bt: [N, Kd] bf16 (pre-transposed weight), so all
// fragments load as two contiguous 16-byte chunks per lane, matching the
// CDNA5 16-bit A-fragment layout:
//   lane l: row = l&15, elems 0..7  -> K = kb + 8*(l>>4) .. +7
//           elems 8..15            -> K = kb + 16 + 8*(l>>4) .. +7
// ACT: 0 = none, 1 = tanh, 2 = sigmoid (compile-time; branch-free epilogue)

template <int MT, int NT, int ACT>
__global__ void gemm_bf16_wmma(const __bf16* __restrict__ A,
                               const __bf16* __restrict__ Bt,
                               const float*  __restrict__ bias,
                               float* __restrict__ C,
                               int M, int N, int Kd) {
    const int wave = (blockIdx.x * blockDim.x + threadIdx.x) >> 5;
    const int lane = threadIdx.x & 31;
    const int nblocks = N / (16 * NT);
    const int mb = wave / nblocks;
    const int nb = wave - mb * nblocks;
    if (mb * MT * 16 >= M) return;         // wave-uniform guard

    const int l15 = lane & 15;
    const int kh  = lane >> 4;             // which K half-group this lane holds

    const __bf16* arow[MT];
    const __bf16* brow[NT];
#pragma unroll
    for (int mi = 0; mi < MT; ++mi)
        arow[mi] = A + (size_t)((mb * MT + mi) * 16 + l15) * Kd + 8 * kh;
#pragma unroll
    for (int ni = 0; ni < NT; ++ni)
        brow[ni] = Bt + (size_t)((nb * NT + ni) * 16 + l15) * Kd + 8 * kh;

    v8f acc[MT][NT];
#pragma unroll
    for (int mi = 0; mi < MT; ++mi)
#pragma unroll
        for (int ni = 0; ni < NT; ++ni)
            acc[mi][ni] = (v8f){};

    for (int kb = 0; kb < Kd; kb += 32) {
        v16bf av[MT], bv[NT];
#pragma unroll
        for (int mi = 0; mi < MT; ++mi) {
            v8bf lo = *(const v8bf*)(arow[mi] + kb);
            v8bf hi = *(const v8bf*)(arow[mi] + kb + 16);
            av[mi] = __builtin_shufflevector(lo, hi,
                        0,1,2,3,4,5,6,7,8,9,10,11,12,13,14,15);
        }
#pragma unroll
        for (int ni = 0; ni < NT; ++ni) {
            v8bf lo = *(const v8bf*)(brow[ni] + kb);
            v8bf hi = *(const v8bf*)(brow[ni] + kb + 16);
            bv[ni] = __builtin_shufflevector(lo, hi,
                        0,1,2,3,4,5,6,7,8,9,10,11,12,13,14,15);
        }
#pragma unroll
        for (int mi = 0; mi < MT; ++mi)
#pragma unroll
            for (int ni = 0; ni < NT; ++ni)
                acc[mi][ni] = __builtin_amdgcn_wmma_f32_16x16x32_bf16(
                    false, av[mi], false, bv[ni],
                    (short)0, acc[mi][ni], false, false);
    }

    // C/D fragment: column = lane&15 (lanes 16-31 alias cols 0-15, rows +8)
#pragma unroll
    for (int ni = 0; ni < NT; ++ni) {
        const int ncol = (nb * NT + ni) * 16 + l15;
        const float bs = bias[ncol];
#pragma unroll
        for (int mi = 0; mi < MT; ++mi) {
#pragma unroll
            for (int r = 0; r < 8; ++r) {
                const int mrow = (mb * MT + mi) * 16 + r + 8 * kh;
                float v = acc[mi][ni][r] + bs;
                if (ACT == 1)      v = fast_tanh(v);
                else if (ACT == 2) v = fast_sigmoid(v);
                C[(size_t)mrow * N + ncol] = v;
            }
        }
    }
}

// -------------------------------- Stage 3 ----------------------------------
// One wave per (b, d) row: lane holds H[k=2*lane], H[k=2*lane+1] in registers
// across all T steps. K-reduction via 5-step shfl_xor butterfly (wave32).
// H_all row store = 32 lanes x float2 = 256 contiguous bytes (coalesced),
// written non-temporally (write-once 1.08 GB stream; keep L2 for operands).

__global__ void scan_kernel(const float* __restrict__ key,
                            const float* __restrict__ dec,
                            const float* __restrict__ val,
                            const float* __restrict__ qry,
                            const float* __restrict__ z,
                            const float* __restrict__ H0,
                            float* __restrict__ out,
                            float* __restrict__ Hall) {
    const int wave = (blockIdx.x * blockDim.x + threadIdx.x) >> 5;
    const int lane = threadIdx.x & 31;
    const int b = wave >> 10;          // 0..7
    const int d = wave & 1023;         // 0..1023
    const int k0 = lane << 1;          // 0,2,..,62

    // Initial state + H_all[0] = H0
    const size_t rowHK = ((size_t)b * D_MODEL + d) * D_STATE + k0;
    v2f h = *(const v2f*)(H0 + rowHK);
    __builtin_nontemporal_store(h, (v2f*)(Hall + rowHK));   // t = 0 slot

    for (int t = 0; t < T_STEPS; ++t) {
        const int rb = t * BATCH + b;
        const size_t rd = (size_t)rb * D_MODEL + d;
        const float kk = key[rd];                       // wave-uniform
        const float dd = dec[rd];
        const float zz = z[rd];
        const v2f qv = *(const v2f*)(qry + (size_t)rb * D_STATE + k0);
        const v2f vv = *(const v2f*)(val + (size_t)rb * D_STATE + k0);

        h.x = dd * h.x + kk * vv.x;
        h.y = dd * h.y + kk * vv.y;

        float p = h.x * qv.x + h.y * qv.y;
#pragma unroll
        for (int off = 16; off > 0; off >>= 1)
            p += __shfl_xor(p, off, 32);

        if (lane == 0) {
            __builtin_nontemporal_store(p * (zz * fast_sigmoid(zz)),
                                        out + rd);      // pre_out * silu(z)
        }
        __builtin_nontemporal_store(h,
            (v2f*)(Hall + ((size_t)(t + 1) * BATCH * D_MODEL
                           + (size_t)b * D_MODEL + d) * D_STATE + k0));
    }
}

// -------------------------------- Launch -----------------------------------

extern "C" void kernel_launch(void* const* d_in, const int* in_sizes, int n_in,
                              void* d_out, int out_size, void* d_ws, size_t ws_size,
                              hipStream_t stream) {
    const float* x  = (const float*)d_in[0];   // [T,B,D]
    const float* z  = (const float*)d_in[1];   // [T,B,D]
    const float* H0 = (const float*)d_in[2];   // [B,D,K]
    const float* Wk = (const float*)d_in[3];   // [D,D]
    const float* bk = (const float*)d_in[4];
    const float* Wv = (const float*)d_in[5];   // [D,K]
    const float* bv = (const float*)d_in[6];
    const float* Wq = (const float*)d_in[7];   // [D,K]
    const float* bq = (const float*)d_in[8];
    const float* Wd = (const float*)d_in[9];   // [D,D]
    const float* bd = (const float*)d_in[10];

    float* out  = (float*)d_out;                               // [T,B,D]
    float* Hall = out + (size_t)T_STEPS * BATCH * D_MODEL;     // [T+1,B,D,K]

    // Workspace carve-up (bytes), total ~48.5 MB
    char* ws = (char*)d_ws;
    size_t off = 0;
    __bf16* xbf = (__bf16*)(ws + off); off += (size_t)M_ROWS * D_MODEL * 2;
    __bf16* wtk = (__bf16*)(ws + off); off += (size_t)D_MODEL * D_MODEL * 2;
    __bf16* wtd = (__bf16*)(ws + off); off += (size_t)D_MODEL * D_MODEL * 2;
    __bf16* wtv = (__bf16*)(ws + off); off += (size_t)D_STATE * D_MODEL * 2;
    __bf16* wtq = (__bf16*)(ws + off); off += (size_t)D_STATE * D_MODEL * 2;
    float*  key = (float*)(ws + off);  off += (size_t)M_ROWS * D_MODEL * 4;
    float*  dec = (float*)(ws + off);  off += (size_t)M_ROWS * D_MODEL * 4;
    float*  val = (float*)(ws + off);  off += (size_t)M_ROWS * D_STATE * 4;
    float*  qry = (float*)(ws + off);  off += (size_t)M_ROWS * D_STATE * 4;
    (void)ws_size; (void)in_sizes; (void)n_in; (void)out_size;

    // Stage 1: conversions
    {
        const int n = M_ROWS * D_MODEL;                       // 4,194,304
        cvt_f32_to_bf16<<<(n + 255) / 256, 256, 0, stream>>>(x, xbf, n);
    }
    {
        const int n = D_MODEL * D_MODEL;                      // 1,048,576
        transpose_cvt_bf16<<<(n + 255) / 256, 256, 0, stream>>>(Wk, wtk, D_MODEL, D_MODEL);
        transpose_cvt_bf16<<<(n + 255) / 256, 256, 0, stream>>>(Wd, wtd, D_MODEL, D_MODEL);
    }
    {
        const int n = D_MODEL * D_STATE;                      // 65,536
        transpose_cvt_bf16<<<(n + 255) / 256, 256, 0, stream>>>(Wv, wtv, D_MODEL, D_STATE);
        transpose_cvt_bf16<<<(n + 255) / 256, 256, 0, stream>>>(Wq, wtq, D_MODEL, D_STATE);
    }

    // Stage 2: WMMA GEMMs.
    // Big GEMMs (N=1024): 32x32 block per wave (2x2 tiles) -> 4096 waves,
    // 4 pinned accumulators keep the WMMA pipeline free of hazard nops.
    {
        const int waves = (M_ROWS / 32) * (D_MODEL / 32);     // 128*32 = 4096
        gemm_bf16_wmma<2, 2, 1><<<waves / 8, 256, 0, stream>>>(
            xbf, wtk, bk, key, M_ROWS, D_MODEL, D_MODEL);
        gemm_bf16_wmma<2, 2, 2><<<waves / 8, 256, 0, stream>>>(
            xbf, wtd, bd, dec, M_ROWS, D_MODEL, D_MODEL);
    }
    // Small GEMMs (N=64): 1x1 tiles for wave-count parallelism -> 1024 waves.
    {
        const int waves = (M_ROWS / 16) * (D_STATE / 16);     // 1024
        gemm_bf16_wmma<1, 1, 0><<<waves / 8, 256, 0, stream>>>(
            xbf, wtv, bv, val, M_ROWS, D_STATE, D_MODEL);
        gemm_bf16_wmma<1, 1, 0><<<waves / 8, 256, 0, stream>>>(
            xbf, wtq, bq, qry, M_ROWS, D_STATE, D_MODEL);
    }

    // Stage 3: scan — 8192 waves, 8 waves per block
    scan_kernel<<<(BATCH * D_MODEL) / 8, 256, 0, stream>>>(
        key, dec, val, qry, z, H0, out, Hall);
}
